// Attention_54692113547782
// MI455X (gfx1250) — matrix operand order repaired and checked by reading, not compiled, not run on hardware
//
#include <hip/hip_runtime.h>
#include <hip/hip_bf16.h>

// Problem constants (fixed by the reference): B=2,H=8 -> BH=16, T=2048, N=1024, D=128
#define BH   16
#define SEQ  2048
#define NDIM 1024
#define DDIM 128
#define BM   64          // query rows per workgroup
#define BK   32          // key rows per tile (double-buffered async stream)

typedef __attribute__((ext_vector_type(16))) __bf16 bf16x16;
typedef __attribute__((ext_vector_type(8)))  __bf16 bf16x8;
typedef __attribute__((ext_vector_type(8)))  float  v8f;

union BF16x16 { bf16x16 v; bf16x8 h[2]; };

#if defined(__AMDGCN__) && __has_builtin(__builtin_amdgcn_global_load_async_to_lds_b128)
#define USE_ASYNC_LDS 1
typedef int i32x4_vec __attribute__((vector_size(16)));
typedef __attribute__((address_space(1))) i32x4_vec gas_i32x4;   // global
typedef __attribute__((address_space(3))) i32x4_vec las_i32x4;   // LDS
#else
#define USE_ASYNC_LDS 0
#endif

__device__ __forceinline__ void async_copy_b128(const void* g, void* l) {
#if USE_ASYNC_LDS
    __builtin_amdgcn_global_load_async_to_lds_b128((gas_i32x4*)g, (las_i32x4*)l, 0, 0);
#else
    *(uint4*)l = *(const uint4*)g;   // fallback: VGPR bounce
#endif
}

__device__ __forceinline__ void wait_async_le16() {
#if USE_ASYNC_LDS
#if __has_builtin(__builtin_amdgcn_s_wait_asynccnt)
    __builtin_amdgcn_s_wait_asynccnt(16);
#else
    asm volatile("s_wait_asynccnt 0x10" ::: "memory");
#endif
#endif
}

__device__ __forceinline__ void wait_async_0() {
#if USE_ASYNC_LDS
#if __has_builtin(__builtin_amdgcn_s_wait_asynccnt)
    __builtin_amdgcn_s_wait_asynccnt(0);
#else
    asm volatile("s_wait_asynccnt 0x0" ::: "memory");
#endif
#endif
}

// ---------------------------------------------------------------------------
// Kernel 1: RoPE + f32 -> bf16 conversion, writes QR to workspace.
// ---------------------------------------------------------------------------
__global__ __launch_bounds__(256)
void rope_bf16_kernel(const float* __restrict__ Q, __bf16* __restrict__ QR) {
    const long long pid = (long long)blockIdx.x * 256 + threadIdx.x; // pair index
    const int NP = NDIM / 2;
    const int  n2  = (int)(pid % NP);
    const long long row = pid / NP;                                  // bh*T + t
    const int  t   = (int)(row % SEQ);

    const float p     = (float)(2 * n2);
    const float freq  = exp2f(-p * (1.0f / 64.0f)) * 0.15915494309189535f; // /(2*pi)
    float phase = (float)t * freq;
    phase -= floorf(phase);
    const float ang = 6.283185307179586f * phase;
    float sn, cs;
    __sincosf(ang, &sn, &cs);

    const long long base = row * NDIM + 2 * n2;
    const float q0 = Q[base];
    const float q1 = Q[base + 1];
    union { __bf16 h[2]; unsigned u; } pk;
    pk.h[0] = (__bf16)(q0 * cs - q1 * sn);
    pk.h[1] = (__bf16)(q1 * cs + q0 * sn);
    *(unsigned*)&QR[base] = pk.u;
}

// ---------------------------------------------------------------------------
// WMMA fragment loaders (CDNA5 wave32 16-bit layouts).
// ---------------------------------------------------------------------------
__device__ __forceinline__
bf16x16 load_a_frag(const __bf16* M, int ld, int row0, int k0, int half, int r) {
    const __bf16* base = M + (long long)(row0 + r) * ld + k0 + half * 8;
    BF16x16 f;
    f.h[0] = *(const bf16x8*)(base);
    f.h[1] = *(const bf16x8*)(base + 16);
    return f.v;
}

__device__ __forceinline__
bf16x16 load_b_frag(const __bf16* M, int ld, int col0, int k0, int half, int r) {
    const __bf16* base = M + (long long)(col0 + r) * ld + k0 + half * 8;
    BF16x16 f;
    f.h[0] = *(const bf16x8*)(base);
    f.h[1] = *(const bf16x8*)(base + 16);
    return f.v;
}

__device__ __forceinline__
v8f wmma_bf16(bf16x16 a, bf16x16 b, v8f c) {
    return __builtin_amdgcn_wmma_f32_16x16x32_bf16(
        false, a, false, b, (short)0, c, false, false);
}

// ---------------------------------------------------------------------------
// Kernel 2: flash-style strict-causal linear attention.
// Grid: (T/BM, BH). Block: 256 threads = 8 waves (wr = wave>>1, wc = wave&1).
// LDS: Qs[64][1024] (128K) | Ks[2][32][1024] (128K) | Vt[128][32] (8K) | Ss[64][32] (4K)
//      = 268 KB of the WGP's 320 KB.
// Key tiles stream through the double buffer via async global->LDS loads.
// ---------------------------------------------------------------------------
#define QS_OFF  0
#define KS_OFF  131072
#define KS_ELEM (BK * NDIM)          // 32768 bf16 per buffer
#define VT_OFF  (KS_OFF + 131072)    // 262144
#define SS_OFF  (VT_OFF + 8192)      // 270336
#define SMEM_BYTES (SS_OFF + 4096)   // 274432

__global__ __launch_bounds__(256)
void attn_rope_kernel(const __bf16* __restrict__ QR,
                      const float*  __restrict__ V,
                      float*        __restrict__ Out) {
    extern __shared__ char smem[];
    __bf16* Qs = (__bf16*)(smem + QS_OFF);
    __bf16* Ks = (__bf16*)(smem + KS_OFF);   // two buffers of KS_ELEM
    __bf16* Vt = (__bf16*)(smem + VT_OFF);   // [d][s] : 128 x 32
    __bf16* Ss = (__bf16*)(smem + SS_OFF);   // [t][s] : 64 x 32

    const int qt  = blockIdx.x;              // query tile 0..31
    const int bh  = blockIdx.y;              // 0..15
    const int t0  = qt * BM;
    const int tid = threadIdx.x;
    const int wave = tid >> 5;
    const int lane = tid & 31;
    const int half = lane >> 4;
    const int r    = lane & 15;
    const int wr   = wave >> 1;              // M tile (16 rows)
    const int wc   = wave & 1;

    // ---- resident QR query tile -> LDS (contiguous 128 KB span) ----
    {
        const uint4* qg = (const uint4*)(QR + ((long long)(bh * SEQ + t0)) * NDIM);
        uint4* ql = (uint4*)Qs;
        #pragma unroll 4
        for (int i = tid; i < (BM * NDIM * 2) / 16; i += 256) ql[i] = qg[i];
    }

    const int last = 2 * qt + 1;             // key tiles kt = 0 .. last (BK=32 rows)

    // ---- prologue: start async stream of key tile 0 into buffer 0 ----
    {
        const __bf16* kg = QR + ((long long)(bh * SEQ)) * NDIM;
        #pragma unroll 4
        for (int i = tid; i < KS_ELEM / 8; i += 256)      // 16 b128 per thread
            async_copy_b128(kg + i * 8, Ks + i * 8);
    }

    v8f oacc[4] = {v8f{}, v8f{}, v8f{}, v8f{}};

    for (int kt = 0; kt <= last; ++kt) {
        const int s0  = kt * BK;
        const int buf = kt & 1;
        __bf16* Kcur = Ks + buf * KS_ELEM;

        // ---- issue async loads for the NEXT key tile into the other buffer
        if (kt < last) {
            const __bf16* kg = QR + ((long long)(bh * SEQ + s0 + BK)) * NDIM;
            __bf16* kl = Ks + (buf ^ 1) * KS_ELEM;
            #pragma unroll 4
            for (int i = tid; i < KS_ELEM / 8; i += 256)
                async_copy_b128(kg + i * 8, kl + i * 8);
        }

        // ---- manual load + convert + transpose of the V tile (small) ----
        {
            const float* vg = V + ((long long)(bh * SEQ + s0)) * DDIM;
            #pragma unroll 4
            for (int i = tid; i < BK * DDIM; i += 256) {
                const int s = i >> 7;           // 0..31
                const int d = i & (DDIM - 1);   // 0..127
                Vt[d * BK + s] = (__bf16)vg[i];
            }
        }

        // current tile's async loads (issued last iteration) are complete once
        // ASYNCcnt drops to the 16 just issued for the next tile (in-order).
        if (kt < last) wait_async_le16(); else wait_async_0();
        __syncthreads();

        // ---- S = Qtile (64xN) x Ktile^T (Nx32); wave -> 16x16 tile, K in 32-steps
        v8f sacc = v8f{};
        #pragma unroll 4
        for (int kk = 0; kk < NDIM; kk += 32) {
            bf16x16 a = load_a_frag(Qs,   NDIM, 16 * wr, kk, half, r);
            bf16x16 b = load_b_frag(Kcur, NDIM, 16 * wc, kk, half, r);
            sacc = wmma_bf16(a, b, sacc);
        }

        // ---- strict causal mask (keep s < t); only tiles on/above t0 ----
        if (kt >= 2 * qt) {
            const int coff = 32 * kt - 64 * qt;   // s_local = coff + c vs t_local = m
            #pragma unroll
            for (int i = 0; i < 8; ++i) {
                const int m = 16 * wr + i + half * 8;
                if (coff + 16 * wc + r >= m) sacc[i] = 0.0f;
            }
        }

        // ---- spill S tile to LDS as bf16 ([t][s], ld = 32) ----
        #pragma unroll
        for (int i = 0; i < 8; ++i) {
            const int m = 16 * wr + i + half * 8;
            Ss[m * BK + 16 * wc + r] = (__bf16)sacc[i];
        }
        __syncthreads();

        // ---- O += S (64x32) x Vtile (32x128); wave owns 16 rows x 64 cols ----
        {
            bf16x16 a = load_a_frag(Ss, BK, 16 * wr, 0, half, r);
            #pragma unroll
            for (int dt = 0; dt < 4; ++dt) {
                bf16x16 b = load_b_frag(Vt, BK, 64 * wc + 16 * dt, 0, half, r);
                oacc[dt] = wmma_bf16(a, b, oacc[dt]);
            }
        }
        __syncthreads();   // all reads of Ss/Vt/Kcur done before next overwrite
    }

    // ---- epilogue: write O accumulators (C layout) to global f32 ----
    float* og = Out + ((long long)(bh * SEQ + t0)) * DDIM;
    #pragma unroll
    for (int dt = 0; dt < 4; ++dt) {
        #pragma unroll
        for (int i = 0; i < 8; ++i) {
            const int m = 16 * wr + i + half * 8;
            const int d = 64 * wc + 16 * dt + r;
            og[m * DDIM + d] = oacc[dt][i];
        }
    }
}

// ---------------------------------------------------------------------------
// Launch: inputs are {Q, K(==Q), V} f32; output (B,H,T,D) f32.
// Workspace holds QR in bf16: 16*2048*1024*2 = 64 MiB.
// ---------------------------------------------------------------------------
extern "C" void kernel_launch(void* const* d_in, const int* in_sizes, int n_in,
                              void* d_out, int out_size, void* d_ws, size_t ws_size,
                              hipStream_t stream) {
    const float* Q = (const float*)d_in[0];
    const float* V = (const float*)d_in[2];
    float* Out = (float*)d_out;
    __bf16* QR = (__bf16*)d_ws;   // 64 MiB of scratch

    const long long pairs = (long long)BH * SEQ * (NDIM / 2);
    rope_bf16_kernel<<<(int)(pairs / 256), 256, 0, stream>>>(Q, QR);

    dim3 grid(SEQ / BM, BH);      // (32, 16)
    attn_rope_kernel<<<grid, 256, SMEM_BYTES, stream>>>(QR, V, Out);
}
